// Continuous_Tucker_ssf_13056700580158
// MI455X (gfx1250) — matrix operand order repaired and checked by reading, not compiled, not run on hardware
//
#include <hip/hip_runtime.h>

typedef _Float16 h16;
typedef __attribute__((ext_vector_type(16))) _Float16 v16h;
typedef __attribute__((ext_vector_type(8)))  _Float16 v8h;
typedef __attribute__((ext_vector_type(8)))  float    v8f;

#define OMEGA 4.0f
#define MID   512
#define TB    64          // batch rows per workgroup
#define H_STRIDE 520      // padded f16 row stride (bank-conflict-free, 16B aligned)
#define NO_STRIDE 33      // padded f32 row stride for net outputs

// ---- workspace layout (f16 elements) ----
#define W2FRAG_PER_NET (512*512)          // 262144
#define W3FRAG_PER_NET (512*32)           // 16384
#define W2FRAG_OFF 0
#define W2FRAG_TOTAL (3*W2FRAG_PER_NET)   // 786432
#define W3FRAG_OFF W2FRAG_TOTAL
#define W3FRAG_TOTAL (3*W3FRAG_PER_NET)   // 49152
#define COREF_OFF (W2FRAG_TOTAL + W3FRAG_TOTAL)  // 835584
#define COREF_TOTAL (1024*32)             // 32768
#define WS_F16_TOTAL (COREF_OFF + COREF_TOTAL)   // 868352

// Fragment order for B (32x16 f16 tile): lane L holds 16 contiguous K values,
//   k = k0 + 16*(L/16) + e,  j = j0 + (L%16)   (contiguous k order, same as the A reads)
// Stored so each lane reads 32 contiguous bytes (2x b128).
__global__ void __launch_bounds__(256)
tucker_convert_kernel(const float* __restrict__ w2u, const float* __restrict__ w2v,
                      const float* __restrict__ w2w,
                      const float* __restrict__ w3u, const float* __restrict__ w3v,
                      const float* __restrict__ w3w,
                      const float* __restrict__ core, h16* __restrict__ ws)
{
    int idx = blockIdx.x * 256 + threadIdx.x;
    if (idx >= WS_F16_TOTAL) return;
    const float* const W2[3] = {w2u, w2v, w2w};
    const float* const W3[3] = {w3u, w3v, w3w};

    if (idx < W2FRAG_TOTAL) {
        // w2 transposed fragments: B[k,j] = w2[j,k], tiles ti = kb*32 + jt
        int net = idx / W2FRAG_PER_NET;
        int rem = idx % W2FRAG_PER_NET;
        int ti  = rem >> 9;          // 16 kb * 32 jt
        int le  = rem & 511;
        int lane = le >> 4, e = le & 15;
        int kb = ti >> 5, jt = ti & 31;
        int k = kb * 32 + ((lane >> 4) << 4) + e;
        int j = jt * 16 + (lane & 15);
        ws[idx] = (h16)W2[net][j * 512 + k];
    } else if (idx < COREF_OFF) {
        // w3 transposed fragments: B[k,j] = w3[j,k], tiles ti = kb*2 + jt
        int rem2 = idx - W3FRAG_OFF;
        int net = rem2 / W3FRAG_PER_NET;
        int rem = rem2 % W3FRAG_PER_NET;
        int ti  = rem >> 9;          // 16 kb * 2 jt
        int le  = rem & 511;
        int lane = le >> 4, e = le & 15;
        int kb = ti >> 1, jt = ti & 1;
        int k = kb * 32 + ((lane >> 4) << 4) + e;
        int j = jt * 16 + (lane & 15);
        ws[idx] = (h16)W3[net][j * 512 + k];
    } else {
        // core as (1024 x 32): B[rs, t] = core[rs*32 + t], tiles ti = kb*2 + jt
        int rem = idx - COREF_OFF;
        int ti  = rem >> 9;          // 32 kb * 2 jt
        int le  = rem & 511;
        int lane = le >> 4, e = le & 15;
        int kb = ti >> 1, jt = ti & 1;
        int k = kb * 32 + ((lane >> 4) << 4) + e;
        int t = jt * 16 + (lane & 15);
        ws[idx] = (h16)core[k * 32 + t];
    }
}

__device__ __forceinline__ v16h make_v16h(v8h lo, v8h hi) {
    return __builtin_shufflevector(lo, hi, 0,1,2,3,4,5,6,7,8,9,10,11,12,13,14,15);
}
__device__ __forceinline__ v8f wmma_f16(v16h a, v16h b, v8f c) {
    return __builtin_amdgcn_wmma_f32_16x16x32_f16(false, a, false, b, (short)0, c,
                                                  false, false);
}
__device__ __forceinline__ float sinsin(float x) { return __sinf(__sinf(x)); }

__global__ void __launch_bounds__(256)
siren_tucker_fused_kernel(const float* __restrict__ tib,
                          const float* __restrict__ w1u, const float* __restrict__ b1u,
                          const float* __restrict__ b2u, const float* __restrict__ b3u,
                          const float* __restrict__ w1v, const float* __restrict__ b1v,
                          const float* __restrict__ b2v, const float* __restrict__ b3v,
                          const float* __restrict__ w1w, const float* __restrict__ b1w,
                          const float* __restrict__ b2w, const float* __restrict__ b3w,
                          const h16* __restrict__ ws, float* __restrict__ out)
{
    extern __shared__ char smem[];
    h16*   h1     = (h16*)smem;                        // TB * H_STRIDE f16
    h16*   h2     = h1 + TB * H_STRIDE;                // TB * H_STRIDE f16
    float* netout = (float*)(h2 + TB * H_STRIDE);      // 3 * TB * NO_STRIDE f32
    float* outacc = netout + 3 * TB * NO_STRIDE;       // TB f32

    const float* const W1[3] = {w1u, w1v, w1w};
    const float* const B1[3] = {b1u, b1v, b1w};
    const float* const B2[3] = {b2u, b2v, b2w};
    const float* const B3[3] = {b3u, b3v, b3w};

    const int tid  = threadIdx.x;
    const int lane = tid & 31;
    const int wave = tid >> 5;
    const int base = blockIdx.x * TB;
    const int lrow = lane & 15;        // lane's row/col within 16
    const int half = lane >> 4;        // 0: lanes 0-15, 1: lanes 16-31

    if (tid < TB) outacc[tid] = 0.0f;

    #pragma unroll
    for (int n = 0; n < 3; ++n) {
        const h16* w2f = ws + W2FRAG_OFF + n * W2FRAG_PER_NET;
        const h16* w3f = ws + W3FRAG_OFF + n * W3FRAG_PER_NET;

        // ---- Layer 1: h1[row,k] = sinsin(omega*(x*w1[k] + b1[k])) ----
        for (int i = tid; i < TB * MID; i += 256) {
            int row = i >> 9, k = i & 511;
            float x = tib[(base + row) * 3 + n];
            h1[row * H_STRIDE + k] =
                (h16)sinsin(OMEGA * (x * W1[n][k] + B1[n][k]));
        }
        __syncthreads();

        // ---- Layer 2: (TB x 512) = sinsin(omega*(h1 @ w2^T + b2)) via WMMA ----
        {
            const int rg = wave >> 1;         // row-group 0..3
            const int jh = wave & 1;          // column half 0..1
            const int arow = rg * 16 + lrow;
            #pragma unroll
            for (int pass = 0; pass < 4; ++pass) {
                const int jt0 = jh * 16 + pass * 4;
                v8f acc[4];
                #pragma unroll
                for (int m = 0; m < 4; ++m) acc[m] = (v8f)0.0f;
                for (int kb = 0; kb < 16; ++kb) {
                    const int koff = kb * 32 + (half << 4);
                    v16h a = make_v16h(*(const v8h*)&h1[arow * H_STRIDE + koff],
                                       *(const v8h*)&h1[arow * H_STRIDE + koff + 8]);
                    if (kb < 15)
                        __builtin_prefetch(w2f + (((kb + 1) * 32 + jt0) << 9) + (lane << 4), 0, 1);
                    #pragma unroll
                    for (int m = 0; m < 4; ++m) {
                        const v8h* bp = (const v8h*)(w2f + (((kb * 32) + jt0 + m) << 9) + (lane << 4));
                        acc[m] = wmma_f16(a, make_v16h(bp[0], bp[1]), acc[m]);
                    }
                }
                #pragma unroll
                for (int m = 0; m < 4; ++m) {
                    const int j = (jt0 + m) * 16 + lrow;
                    const float bias = B2[n][j];
                    #pragma unroll
                    for (int v = 0; v < 8; ++v) {
                        const int row = rg * 16 + v + (half << 3);
                        h2[row * H_STRIDE + j] =
                            (h16)sinsin(OMEGA * (acc[m][v] + bias));
                    }
                }
            }
        }
        __syncthreads();

        // ---- Layer 3: (TB x 32) = h2 @ w3^T + b3, one 16x16 tile per wave ----
        {
            const int rg = wave >> 1;
            const int jt = wave & 1;
            const int arow = rg * 16 + lrow;
            v8f acc = (v8f)0.0f;
            for (int kb = 0; kb < 16; ++kb) {
                const int koff = kb * 32 + (half << 4);
                v16h a = make_v16h(*(const v8h*)&h2[arow * H_STRIDE + koff],
                                   *(const v8h*)&h2[arow * H_STRIDE + koff + 8]);
                const v8h* bp = (const v8h*)(w3f + ((kb * 2 + jt) << 9) + (lane << 4));
                acc = wmma_f16(a, make_v16h(bp[0], bp[1]), acc);
            }
            const int j = jt * 16 + lrow;
            const float bias = B3[n][j];
            #pragma unroll
            for (int v = 0; v < 8; ++v) {
                const int row = rg * 16 + v + (half << 3);
                netout[n * TB * NO_STRIDE + row * NO_STRIDE + j] = acc[v] + bias;
            }
        }
        __syncthreads();
    }

    // ---- Tucker: out[b] = (G @ C2)[b,:] . W[b,:],  G[b, r*32+s] = U[b,r]*V[b,s]
    // A-fragments of G built on the fly: per K-block r, a = U[b,r] * Vfrag.
    {
        const h16* coref = ws + COREF_OFF;
        const int rg = wave >> 1;
        const int jt = wave & 1;
        const int arow = rg * 16 + lrow;

        // V fragment (A layout, f32): element e -> s = e + 16*half
        const float* Vrow = &netout[1 * TB * NO_STRIDE + arow * NO_STRIDE];
        const float* Urow = &netout[0 * TB * NO_STRIDE + arow * NO_STRIDE];
        float vv[16];
        #pragma unroll
        for (int e = 0; e < 16; ++e)
            vv[e] = Vrow[e + (half << 4)];

        v8f acc = (v8f)0.0f;
        for (int r = 0; r < 32; ++r) {
            const float u = Urow[r];
            v16h a;
            #pragma unroll
            for (int e = 0; e < 16; ++e) a[e] = (h16)(u * vv[e]);
            const v8h* bp = (const v8h*)(coref + ((r * 2 + jt) << 9) + (lane << 4));
            acc = wmma_f16(a, make_v16h(bp[0], bp[1]), acc);
        }

        // out[b] += sum_t P[b,t] * W[b,t] for this tile's 16 t-columns
        const int t = jt * 16 + lrow;
        #pragma unroll
        for (int v = 0; v < 8; ++v) {
            const int prow = rg * 16 + v + (half << 3);
            float p = acc[v] * netout[2 * TB * NO_STRIDE + prow * NO_STRIDE + t];
            p += __shfl_xor(p, 1, 32);
            p += __shfl_xor(p, 2, 32);
            p += __shfl_xor(p, 4, 32);
            p += __shfl_xor(p, 8, 32);
            if (lrow == 0) atomicAdd(&outacc[prow], p);
        }
    }
    __syncthreads();
    if (tid < TB) out[base + tid] = outacc[tid];
}

extern "C" void kernel_launch(void* const* d_in, const int* in_sizes, int n_in,
                              void* d_out, int out_size, void* d_ws, size_t ws_size,
                              hipStream_t stream) {
    const float* tib = (const float*)d_in[0];
    // per-net inputs: base 1 + 6*n : w1, b1, w2, b2, w3, b3
    const float* w1u = (const float*)d_in[1];  const float* b1u = (const float*)d_in[2];
    const float* w2u = (const float*)d_in[3];  const float* b2u = (const float*)d_in[4];
    const float* w3u = (const float*)d_in[5];  const float* b3u = (const float*)d_in[6];
    const float* w1v = (const float*)d_in[7];  const float* b1v = (const float*)d_in[8];
    const float* w2v = (const float*)d_in[9];  const float* b2v = (const float*)d_in[10];
    const float* w3v = (const float*)d_in[11]; const float* b3v = (const float*)d_in[12];
    const float* w1w = (const float*)d_in[13]; const float* b1w = (const float*)d_in[14];
    const float* w2w = (const float*)d_in[15]; const float* b2w = (const float*)d_in[16];
    const float* w3w = (const float*)d_in[17]; const float* b3w = (const float*)d_in[18];
    const float* core = (const float*)d_in[19];
    float* out = (float*)d_out;
    h16* ws = (h16*)d_ws;

    const int conv_blocks = (WS_F16_TOTAL + 255) / 256;
    tucker_convert_kernel<<<conv_blocks, 256, 0, stream>>>(
        w2u, w2v, w2w, w3u, w3v, w3w, core, ws);

    const size_t lds_bytes = (size_t)(2 * TB * H_STRIDE) * sizeof(h16)
                           + (size_t)(3 * TB * NO_STRIDE + TB) * sizeof(float);
    const int grid = 65536 / TB;   // 1024 workgroups
    siren_tucker_fused_kernel<<<grid, 256, lds_bytes, stream>>>(
        tib,
        w1u, b1u, b2u, b3u,
        w1v, b1v, b2v, b3v,
        w1w, b1w, b2w, b3w,
        ws, out);
}
